// MultiBranchAttention_75368086110523
// MI455X (gfx1250) — matrix operand order repaired and checked by reading, not compiled, not run on hardware
//
#include <hip/hip_runtime.h>
#include <hip/hip_bf16.h>

typedef __attribute__((ext_vector_type(16))) __bf16 v16bf;
typedef __attribute__((ext_vector_type(8)))  __bf16 v8bf;
typedef __attribute__((ext_vector_type(4)))  __bf16 v4bf;
typedef __attribute__((ext_vector_type(8)))  float  v8f;

#define WMMA_BF16(A, B, C) \
  __builtin_amdgcn_wmma_f32_16x16x32_bf16(false, (A), false, (B), (short)0, (C), false, false)

#define CAT16(a, b) \
  __builtin_shufflevector((a), (b), 0, 1, 2, 3, 4, 5, 6, 7, 8, 9, 10, 11, 12, 13, 14, 15)

// ---- CDNA5 async memory->LDS copy (ASYNCcnt-tracked, no VGPR staging) ------
__device__ __forceinline__ void async_load_b128(const void* gptr, const void* lptr) {
  unsigned lds_off = (unsigned)(uintptr_t)lptr;           // LDS aperture: addr[31:0]
  unsigned long long ga = (unsigned long long)(uintptr_t)gptr;
  asm volatile("global_load_async_to_lds_b128 %0, %1, off"
               :: "v"(lds_off), "v"(ga) : "memory");
}
__device__ __forceinline__ void wait_async0() {
  asm volatile("s_wait_asynccnt 0x0" ::: "memory");
}

// ---------------------------------------------------------------------------
// Kernel 0: fp32 -> bf16 bulk convert (one-time, so GEMM loops are cvt-free)
// ---------------------------------------------------------------------------
__global__ void mba_cvt_kernel(const float* __restrict__ src,
                               __bf16* __restrict__ dst, int n4) {
  const int i = blockIdx.x * blockDim.x + threadIdx.x;
  if (i < n4) {
    const float4 v = ((const float4*)src)[i];
    v4bf o;
    o[0] = (__bf16)v.x; o[1] = (__bf16)v.y;
    o[2] = (__bf16)v.z; o[3] = (__bf16)v.w;
    ((v4bf*)dst)[i] = o;
  }
}

// ---------------------------------------------------------------------------
// Kernel 1: qkv = xb @ wqkvb.T  (M=4096, N=1536, K=512), all-bf16 inputs.
//   Q,K head-major [bh][l][d];  V TRANSPOSED per head [bh][d][l]
// ---------------------------------------------------------------------------
__global__ void mba_qkv_kernel(const __bf16* __restrict__ xb,
                               const __bf16* __restrict__ wqkvb,
                               __bf16* __restrict__ Qb,
                               __bf16* __restrict__ Kb,
                               __bf16* __restrict__ Vt) {
  const int lane = threadIdx.x & 31;
  const int wv   = (blockIdx.x * (blockDim.x >> 5)) + (threadIdx.x >> 5);
  const int lm   = lane & 15;
  const int half = lane >> 4;
  const int lo   = half * 8;

  const int NT = 1536 / 64;                // 24 n-tiles of 64
  const int mt = wv / NT;
  const int nt = wv % NT;
  const int m0 = mt * 16;
  const int n0 = nt * 64;
  const int row = m0 + lm;

  v8f acc[4] = {v8f{}, v8f{}, v8f{}, v8f{}};

  for (int k0 = 0; k0 < 512; k0 += 32) {
    const __bf16* ar = xb + (size_t)row * 512 + k0;
    v8bf a0 = *(const v8bf*)(ar + half * 16);
    v8bf a1 = *(const v8bf*)(ar + half * 16 + 8);
    v16bf a = CAT16(a0, a1);

    v16bf bfr[4];
#pragma unroll
    for (int t = 0; t < 4; ++t) {
      const int n = n0 + t * 16 + lm;
      bfr[t] = *(const v16bf*)(wqkvb + (size_t)n * 512 + k0 + half * 16);
    }
#pragma unroll
    for (int t = 0; t < 4; ++t)
      acc[t] = WMMA_BF16(a, bfr[t], acc[t]);
  }

#pragma unroll
  for (int t = 0; t < 4; ++t) {
#pragma unroll
    for (int r = 0; r < 8; ++r) {
      const int row_out = m0 + r + half * 8;          // token index
      const int col     = n0 + t * 16 + lm;           // 0..1535
      const int b  = row_out >> 10;
      const int l  = row_out & 1023;
      const int seg = col >> 9;                       // 0=Q,1=K,2=V
      const int c   = col & 511;
      const int h   = c >> 6;
      const int d   = c & 63;
      const int bh  = b * 8 + h;
      const __bf16 val = (__bf16)acc[t][r];
      if (seg == 0)       Qb[((size_t)bh * 1024 + l) * 64 + d] = val;
      else if (seg == 1)  Kb[((size_t)bh * 1024 + l) * 64 + d] = val;
      else                Vt[((size_t)bh * 64 + d) * 1024 + l] = val;   // transposed
    }
  }
}

// ---------------------------------------------------------------------------
// Kernel 2: fused 3-branch flash attention.
// grid: (B*H) * (L/128) blocks, 256 threads (8 waves x 16 query rows).
// ---------------------------------------------------------------------------
__global__ void mba_attn_kernel(const __bf16* __restrict__ Qb,
                                const __bf16* __restrict__ Kb,
                                const __bf16* __restrict__ Vt,
                                const float* __restrict__ adj,
                                const float* __restrict__ bias_scale,
                                const float* __restrict__ beta,
                                __bf16* __restrict__ Hb) {
  __shared__ __align__(32) __bf16 sK[64 * 64];        // [key][d]
  __shared__ __align__(32) __bf16 sVt[64 * 64];       // [d][key]
  __shared__ __align__(32) __bf16 sP[8][16 * 64];     // per-wave C->A scratch

  const int tid  = threadIdx.x;
  const int wv   = tid >> 5;
  const int lane = tid & 31;
  const int lm   = lane & 15;
  const int half = lane >> 4;
  const int lo   = half * 8;

  const int bh = blockIdx.x >> 3;           // L/128 = 8 q-tiles
  const int qt = blockIdx.x & 7;
  const int b  = bh >> 3;
  const int h  = bh & 7;
  const int q0 = qt * 128 + wv * 16;

  // Q A-fragments for d-chunks [0,32) and [32,64)
  v16bf qf[2];
  const __bf16* Qrow = Qb + ((size_t)bh * 1024 + q0 + lm) * 64;
#pragma unroll
  for (int dk = 0; dk < 2; ++dk) {
    v8bf q0v = *(const v8bf*)(Qrow + dk * 32 + half * 16);
    v8bf q1v = *(const v8bf*)(Qrow + dk * 32 + half * 16 + 8);
    qf[dk] = CAT16(q0v, q1v);
  }

  float mrun[3][8], lrun[3][8];
  v8f o[3][4];
#pragma unroll
  for (int br = 0; br < 3; ++br) {
#pragma unroll
    for (int r = 0; r < 8; ++r) { mrun[br][r] = -1e30f; lrun[br][r] = 0.f; }
#pragma unroll
    for (int n = 0; n < 4; ++n) o[br][n] = v8f{};
  }

  const float inv_scale = 0.125f;           // 1/sqrt(64)
  const float bs0 = bias_scale[0];
  const float bs1 = bias_scale[1];

  for (int kt = 0; kt < 16; ++kt) {
    const int k0 = kt * 64;
    __syncthreads();                        // previous tile fully consumed
    {
      // K tile: contiguous 8 KB region, key-major
      const char* gK = (const char*)(Kb + ((size_t)bh * 1024 + k0) * 64);
      char* lK = (char*)sK;
#pragma unroll
      for (int it = 0; it < 2; ++it) {
        const int i = tid + it * 256;       // 512 x b128
        async_load_b128(gK + i * 16, lK + i * 16);
      }
      // V tile: 64 d-rows x 64 keys from transposed V (row stride 1024)
      const __bf16* gV = Vt + (size_t)bh * 64 * 1024 + k0;
#pragma unroll
      for (int it = 0; it < 2; ++it) {
        const int i  = tid + it * 256;      // 512 x b128
        const int rd = i >> 3;              // d row
        const int ch = i & 7;               // 8-key chunk
        async_load_b128(gV + (size_t)rd * 1024 + ch * 8,
                        sVt + rd * 64 + ch * 8);
      }
      wait_async0();
    }
    __syncthreads();
    if (kt + 1 < 16)
      __builtin_prefetch(Kb + ((size_t)bh * 1024 + k0 + 64) * 64, 0, 1);

    // ---- S = Q K^T for this wave's 16 rows x 64 keys ----
    v8f sacc[4] = {v8f{}, v8f{}, v8f{}, v8f{}};
#pragma unroll
    for (int j = 0; j < 4; ++j) {
#pragma unroll
      for (int dk = 0; dk < 2; ++dk) {
        const int key = j * 16 + lm;
        const int off = key * 64 + dk * 32 + half * 16;   // 32B aligned
        v16bf kf = *(const v16bf*)(sK + off);
        sacc[j] = WMMA_BF16(qf[dk], kf, sacc[j]);
      }
    }
#pragma unroll
    for (int j = 0; j < 4; ++j)
#pragma unroll
      for (int r = 0; r < 8; ++r) sacc[j][r] *= inv_scale;

    // ---- three biased online softmaxes + PV ----
#pragma unroll
    for (int br = 0; br < 3; ++br) {
      float p[4][8];
#pragma unroll
      for (int j = 0; j < 4; ++j) {
        const int kcol = k0 + j * 16 + lm;
        if (br == 0) {
#pragma unroll
          for (int r = 0; r < 8; ++r) p[j][r] = sacc[j][r];
        } else if (br == 1) {
          // forward bias: adj[b][qrow][kcol], strided gather over rows
#pragma unroll
          for (int r = 0; r < 8; ++r) {
            const int qrow = q0 + r + half * 8;
            float bias = 0.f;
            if (qrow != 0 && kcol != 0)
              bias = bs0 * adj[((size_t)b * 1024 + qrow) * 1024 + kcol];
            p[j][r] = sacc[j][r] + bias;
          }
        } else {
          // backward bias: adj[b][kcol][qrow], contiguous in qrow -> 2x float4
          float bias8[8] = {0.f, 0.f, 0.f, 0.f, 0.f, 0.f, 0.f, 0.f};
          if (kcol != 0) {
            const float4* ar = (const float4*)(adj +
                ((size_t)b * 1024 + kcol) * 1024 + q0 + half * 8);
            const float4 a0v = ar[0], a1v = ar[1];
            bias8[0] = a0v.x; bias8[1] = a0v.y; bias8[2] = a0v.z; bias8[3] = a0v.w;
            bias8[4] = a1v.x; bias8[5] = a1v.y; bias8[6] = a1v.z; bias8[7] = a1v.w;
          }
#pragma unroll
          for (int r = 0; r < 8; ++r) {
            const int qrow = q0 + r + half * 8;
            p[j][r] = sacc[j][r] + ((qrow != 0) ? bs1 * bias8[r] : 0.f);
          }
        }
      }
      // row max across 64 keys (4 tiles + 16-lane halves)
      float mx[8];
#pragma unroll
      for (int r = 0; r < 8; ++r)
        mx[r] = fmaxf(fmaxf(p[0][r], p[1][r]), fmaxf(p[2][r], p[3][r]));
#pragma unroll
      for (int off = 1; off < 16; off <<= 1)
#pragma unroll
        for (int r = 0; r < 8; ++r)
          mx[r] = fmaxf(mx[r], __shfl_xor(mx[r], off, 32));

      float corr[8];
#pragma unroll
      for (int r = 0; r < 8; ++r) {
        const float mn = fmaxf(mrun[br][r], mx[r]);
        corr[r] = __expf(mrun[br][r] - mn);
        mrun[br][r] = mn;
      }
#pragma unroll
      for (int j = 0; j < 4; ++j)
#pragma unroll
        for (int r = 0; r < 8; ++r)
          p[j][r] = __expf(p[j][r] - mrun[br][r]);

      float rs[8];
#pragma unroll
      for (int r = 0; r < 8; ++r)
        rs[r] = p[0][r] + p[1][r] + p[2][r] + p[3][r];
#pragma unroll
      for (int off = 1; off < 16; off <<= 1)
#pragma unroll
        for (int r = 0; r < 8; ++r)
          rs[r] += __shfl_xor(rs[r], off, 32);
#pragma unroll
      for (int r = 0; r < 8; ++r)
        lrun[br][r] = lrun[br][r] * corr[r] + rs[r];
#pragma unroll
      for (int n = 0; n < 4; ++n)
#pragma unroll
        for (int r = 0; r < 8; ++r)
          o[br][n][r] *= corr[r];

      // C-layout -> A-layout via per-wave LDS scratch (intra-wave only,
      // backend inserts dscnt waits for same-wave RAW)
      __bf16* pw = sP[wv];
#pragma unroll
      for (int j = 0; j < 4; ++j)
#pragma unroll
        for (int r = 0; r < 8; ++r)
          pw[(r + half * 8) * 64 + j * 16 + lm] = (__bf16)p[j][r];

#pragma unroll
      for (int ks = 0; ks < 2; ++ks) {
        const __bf16* prow = pw + lm * 64 + ks * 32;
        v8bf p0v = *(const v8bf*)(prow + half * 16);
        v8bf p1v = *(const v8bf*)(prow + half * 16 + 8);
        v16bf pa = CAT16(p0v, p1v);
#pragma unroll
        for (int n = 0; n < 4; ++n) {
          const int d = n * 16 + lm;
          // single 32B vector LDS read thanks to d-major sVt
          v16bf vfrag = *(const v16bf*)(sVt + d * 64 + ks * 32 + half * 16);
          o[br][n] = WMMA_BF16(pa, vfrag, o[br][n]);
        }
      }
    }
  }

  // branch weights = softmax(beta)
  const float e0 = __expf(beta[0]);
  const float e1 = __expf(beta[1]);
  const float e2 = __expf(beta[2]);
  const float inv = 1.f / (e0 + e1 + e2);
  const float bw0 = e0 * inv, bw1 = e1 * inv, bw2 = e2 * inv;

#pragma unroll
  for (int n = 0; n < 4; ++n) {
#pragma unroll
    for (int r = 0; r < 8; ++r) {
      const int qrow = q0 + r + half * 8;
      const int d    = n * 16 + lm;
      const float val = bw0 * o[0][n][r] / lrun[0][r]
                      + bw1 * o[1][n][r] / lrun[1][r]
                      + bw2 * o[2][n][r] / lrun[2][r];
      Hb[((size_t)b * 1024 + qrow) * 512 + h * 64 + d] = (__bf16)val;
    }
  }
}

// ---------------------------------------------------------------------------
// Kernel 3: out = Hb @ wpb.T  (M=4096, N=512, K=512), fp32 output
// ---------------------------------------------------------------------------
__global__ void mba_proj_kernel(const __bf16* __restrict__ Hb,
                                const __bf16* __restrict__ wpb,
                                float* __restrict__ out) {
  const int lane = threadIdx.x & 31;
  const int wv   = (blockIdx.x * (blockDim.x >> 5)) + (threadIdx.x >> 5);
  const int lm   = lane & 15;
  const int half = lane >> 4;
  const int lo   = half * 8;

  const int NT = 512 / 64;                  // 8 n-tiles of 64
  const int mt = wv / NT;
  const int nt = wv % NT;
  const int m0 = mt * 16;
  const int n0 = nt * 64;
  const int row = m0 + lm;

  v8f acc[4] = {v8f{}, v8f{}, v8f{}, v8f{}};

  for (int k0 = 0; k0 < 512; k0 += 32) {
    const __bf16* ar = Hb + (size_t)row * 512 + k0;
    v8bf a0 = *(const v8bf*)(ar + half * 16);
    v8bf a1 = *(const v8bf*)(ar + half * 16 + 8);
    v16bf a = CAT16(a0, a1);

    v16bf bfr[4];
#pragma unroll
    for (int t = 0; t < 4; ++t) {
      const int n = n0 + t * 16 + lm;
      bfr[t] = *(const v16bf*)(wpb + (size_t)n * 512 + k0 + half * 16);
    }
#pragma unroll
    for (int t = 0; t < 4; ++t)
      acc[t] = WMMA_BF16(a, bfr[t], acc[t]);
  }

#pragma unroll
  for (int t = 0; t < 4; ++t)
#pragma unroll
    for (int r = 0; r < 8; ++r) {
      const int row_out = m0 + r + half * 8;
      const int col     = n0 + t * 16 + lm;
      out[(size_t)row_out * 512 + col] = acc[t][r];
    }
}

// ---------------------------------------------------------------------------
extern "C" void kernel_launch(void* const* d_in, const int* in_sizes, int n_in,
                              void* d_out, int out_size, void* d_ws, size_t ws_size,
                              hipStream_t stream) {
  const float* x    = (const float*)d_in[0];   // (4,1024,512)
  const float* adj  = (const float*)d_in[1];   // (4,1024,1024)
  const float* Wqkv = (const float*)d_in[2];   // (1536,512)
  const float* Wp   = (const float*)d_in[3];   // (512,512)
  const float* bsc  = (const float*)d_in[4];   // (2,)
  const float* beta = (const float*)d_in[5];   // (3,)
  float* out = (float*)d_out;                  // (4,1024,512) fp32

  const size_t qn  = (size_t)4 * 8 * 1024 * 64;  // 2,097,152 (= B*L*C)
  const size_t wqn = (size_t)1536 * 512;         //   786,432
  const size_t wpn = (size_t)512 * 512;          //   262,144

  __bf16* xb    = (__bf16*)d_ws;
  __bf16* wqkvb = xb + qn;
  __bf16* wpb   = wqkvb + wqn;
  __bf16* Qb    = wpb + wpn;
  __bf16* Kb    = Qb + qn;
  __bf16* Vt    = Kb + qn;                       // transposed per head [bh][d][l]
  __bf16* Hb    = Vt + qn;

  // one-time fp32 -> bf16 conversions (keeps GEMM loops cvt-free)
  mba_cvt_kernel<<<(int)(qn / 4 / 256),  256, 0, stream>>>(x, xb, (int)(qn / 4));
  mba_cvt_kernel<<<(int)(wqn / 4 / 256), 256, 0, stream>>>(Wqkv, wqkvb, (int)(wqn / 4));
  mba_cvt_kernel<<<(int)(wpn / 4 / 256), 256, 0, stream>>>(Wp, wpb, (int)(wpn / 4));

  // 4096/16 * 1536/64 = 6144 waves -> 768 blocks of 8 waves
  mba_qkv_kernel<<<768, 256, 0, stream>>>(xb, wqkvb, Qb, Kb, Vt);
  // B*H * L/128 = 32 * 8 = 256 blocks
  mba_attn_kernel<<<256, 256, 0, stream>>>(Qb, Kb, Vt, adj, bsc, beta, Hb);
  // 4096/16 * 512/64 = 2048 waves -> 256 blocks
  mba_proj_kernel<<<256, 256, 0, stream>>>(Hb, wpb, out);
}
